// RAGEnhancedGNN_48430051230177
// MI455X (gfx1250) — compile-verified, matrix operand order checked by reading, not run on hardware
//
#include <hip/hip_runtime.h>
#include <hip/hip_bf16.h>

typedef __attribute__((ext_vector_type(16))) _Float16 v16h;
typedef __attribute__((ext_vector_type(8)))  float    v8f;

#define N_NODES 50000
#define D_HID   128

__device__ __forceinline__ _Float16 f2h(float f) { return (_Float16)f; }

#if defined(__has_builtin)
#if __has_builtin(__builtin_amdgcn_sched_group_barrier)
#define SCHED_GROUP(mask, size, id) __builtin_amdgcn_sched_group_barrier((mask), (size), (id))
#endif
#endif
#ifndef SCHED_GROUP
#define SCHED_GROUP(mask, size, id)
#endif

// ---------------------------------------------------------------------------
// Degree / normalization kernels
// ---------------------------------------------------------------------------
__global__ void deg_init_kernel(float* deg, int n) {
    int i = blockIdx.x * blockDim.x + threadIdx.x;
    if (i < n) deg[i] = 1.0f;                      // self loop contributes 1
}

__global__ void deg_edge_kernel(const long long* __restrict__ dst, float* deg, int nE) {
    int e = blockIdx.x * blockDim.x + threadIdx.x;
    if (e < nE) atomicAdd(&deg[dst[e]], 1.0f);
}

__global__ void dinv_kernel(float* deg, int n) {
    int i = blockIdx.x * blockDim.x + threadIdx.x;
    if (i < n) {
        float d = deg[i];
        deg[i] = (d > 0.0f) ? rsqrtf(d) : 0.0f;    // in-place deg -> dinv
    }
}

// ---------------------------------------------------------------------------
// WMMA GEMM:  Y[M,N] = X[M,K] @ W[K,N]  (+ bias), f32 in/out, f16 WMMA core.
// Block = 256 threads = 8 waves; wave w computes rows [tile*16, tile*16+16).
// W is staged in LDS as f16, pre-swizzled into per-(kt,ntile,lane) 32B frags.
// sched_group_barrier pins the per-K-step pipeline:
//   [16 DS reads (all B frags)] -> [4 VMEM reads (A)] -> [8 VALU cvt] ->
//   [8 back-to-back WMMA], so iteration k+1's loads issue under iteration
//   k's WMMA chain instead of each WMMA stalling on a fresh LDS round-trip.
// M must be a multiple of 16 (50000 = 3125*16).
// ---------------------------------------------------------------------------
template <int K, int NT, bool BIAS>
__global__ __launch_bounds__(256)
void gcn_gemm_kernel(const float* __restrict__ X, const float* __restrict__ W,
                     const float* __restrict__ bias, float* __restrict__ Y, int M) {
    constexpr int N  = NT * 16;
    constexpr int KT = K / 32;
    __shared__ __align__(32) _Float16 sW[K * N];   // fragment-swizzled layout

    // Cooperative load + swizzle of W into LDS (f32 -> f16).
    for (int idx = threadIdx.x; idx < K * N; idx += blockDim.x) {
        int k = idx / N, n = idx % N;
        int kt = k >> 5, kk = k & 31;
        int laneHi = (kk >> 3) & 1;                     // K 8..15 / 24..31 -> hi half-wave
        int e      = ((kk & 16) >> 1) | (kk & 7);       // element slot 0..15
        int lane   = (n & 15) | (laneHi << 4);
        int nt     = n >> 4;
        sW[((((kt * NT + nt) << 5) + lane) << 4) + e] = f2h(W[idx]);
    }
    __syncthreads();

    const int wave = threadIdx.x >> 5;
    const int lane = threadIdx.x & 31;
    const int tile = blockIdx.x * 8 + wave;
    if (tile * 16 >= M) return;                    // wave-uniform: EXEC stays all-1s

    const int   row = tile * 16 + (lane & 15);
    const float* xr = X + (size_t)row * K;
    const int   kHi = (lane >> 4) << 3;            // 0 or 8

    v8f acc[NT];
#pragma unroll
    for (int nt = 0; nt < NT; ++nt)
        acc[nt] = v8f{0.f, 0.f, 0.f, 0.f, 0.f, 0.f, 0.f, 0.f};

#pragma unroll 2
    for (int kt = 0; kt < KT; ++kt) {
        const int k0 = (kt << 5) + kHi;
        if (kt + 1 < KT) __builtin_prefetch(xr + k0 + 32, 0, 3);  // global_prefetch_b8

        // All NT B fragments for this K-step: keep live in registers.
        v16h b[NT];
#pragma unroll
        for (int nt = 0; nt < NT; ++nt)
            b[nt] = *(const v16h*)(&sW[(((kt * NT + nt) << 5) + lane) << 4]);

        // A fragment: 16-bit A 16x32 layout (lanes 0-15: K=k0..k0+7,k0+16..k0+23)
        float4 p0 = *(const float4*)(xr + k0);
        float4 p1 = *(const float4*)(xr + k0 + 4);
        float4 p2 = *(const float4*)(xr + k0 + 16);
        float4 p3 = *(const float4*)(xr + k0 + 20);
        v16h a;
        a[0]  = f2h(p0.x); a[1]  = f2h(p0.y); a[2]  = f2h(p0.z); a[3]  = f2h(p0.w);
        a[4]  = f2h(p1.x); a[5]  = f2h(p1.y); a[6]  = f2h(p1.z); a[7]  = f2h(p1.w);
        a[8]  = f2h(p2.x); a[9]  = f2h(p2.y); a[10] = f2h(p2.z); a[11] = f2h(p2.w);
        a[12] = f2h(p3.x); a[13] = f2h(p3.y); a[14] = f2h(p3.z); a[15] = f2h(p3.w);

#pragma unroll
        for (int nt = 0; nt < NT; ++nt)
            acc[nt] = __builtin_amdgcn_wmma_f32_16x16x32_f16(
                /*neg_a=*/false, a, /*neg_b=*/false, b[nt],
                /*c_mod=*/(short)0, acc[nt], /*reuse_a=*/false, /*reuse_b=*/false);

        // Pin the schedule: DS reads, VMEM reads, cvt VALU, then WMMA chain.
        SCHED_GROUP(0x100, 2 * NT, 0);   // DS reads   (B fragments)
        SCHED_GROUP(0x020, 4, 0);        // VMEM reads (A float4s)
        SCHED_GROUP(0x002, 8, 0);        // VALU       (f32->f16 packs)
        SCHED_GROUP(0x008, NT, 0);       // WMMA back-to-back
    }

    // C/D layout: VGPR r holds M=r (lanes 0-15) / M=r+8 (lanes 16-31), col = lane&15.
    const int mBase = tile * 16 + kHi;
    const int n0    = lane & 15;
#pragma unroll
    for (int nt = 0; nt < NT; ++nt) {
        float bv = BIAS ? bias[nt * 16 + n0] : 0.0f;
#pragma unroll
        for (int r = 0; r < 8; ++r)
            Y[(size_t)(mBase + r) * N + nt * 16 + n0] = acc[nt][r] + bv;
    }
}

// ---------------------------------------------------------------------------
// Aggregation: out = D^-1/2 (A+I) D^-1/2 h
// ---------------------------------------------------------------------------
// Self-loop contribution without atomics: out[n,:] = h[n,:] * dinv[n]^2
__global__ void agg_init_kernel(const float* __restrict__ h, const float* __restrict__ dinv,
                                float* __restrict__ out, int n, int D) {
    int i = blockIdx.x * blockDim.x + threadIdx.x;          // over n*D/4 float4s
    int total = n * (D >> 2);
    if (i >= total) return;
    int node = i / (D >> 2);
    float w = dinv[node] * dinv[node];
    float4 v = ((const float4*)h)[i];
    ((float4*)out)[i] = float4{v.x * w, v.y * w, v.z * w, v.w * w};
}

// One wave per edge; each lane owns 4 consecutive features (D=128 -> 1 iter).
__global__ void agg_edge_kernel(const float* __restrict__ h,
                                const long long* __restrict__ src,
                                const long long* __restrict__ dst,
                                const float* __restrict__ dinv,
                                float* __restrict__ out, int nE, int D) {
    int e = (blockIdx.x * blockDim.x + threadIdx.x) >> 5;
    if (e >= nE) return;
    int lane = threadIdx.x & 31;
    long long s = src[e], d = dst[e];
    float w = dinv[s] * dinv[d];
    const float4* hp = (const float4*)(h + (size_t)s * D);
    float*        op = out + (size_t)d * D;
    for (int j = lane; j < (D >> 2); j += 32) {
        float4 v = hp[j];
        atomicAdd(op + 4 * j + 0, v.x * w);
        atomicAdd(op + 4 * j + 1, v.y * w);
        atomicAdd(op + 4 * j + 2, v.z * w);
        atomicAdd(op + 4 * j + 3, v.w * w);
    }
}

__global__ void bias_relu_kernel(float* __restrict__ h, const float* __restrict__ b,
                                 int total, int D) {
    int i = blockIdx.x * blockDim.x + threadIdx.x;
    if (i >= total) return;
    float v = h[i] + b[i & (D - 1)];
    h[i] = v > 0.0f ? v : 0.0f;
}

// ---------------------------------------------------------------------------
extern "C" void kernel_launch(void* const* d_in, const int* in_sizes, int n_in,
                              void* d_out, int out_size, void* d_ws, size_t ws_size,
                              hipStream_t stream) {
    const float*     x   = (const float*)d_in[0];
    const long long* ei  = (const long long*)d_in[1];   // int64 [2, E]
    const float*     W1  = (const float*)d_in[2];
    const float*     b1  = (const float*)d_in[3];
    const float*     W2  = (const float*)d_in[4];
    const float*     b2  = (const float*)d_in[5];
    const float*     Wf  = (const float*)d_in[6];
    const float*     bf  = (const float*)d_in[7];
    float*           out = (float*)d_out;

    const int N = N_NODES;
    const int E = in_sizes[1] / 2;
    const long long* src = ei;
    const long long* dst = ei + E;

    // Workspace layout: dinv[N] | t1[N*128] | t2[N*128]
    char*  ws   = (char*)d_ws;
    float* dinv = (float*)ws;
    float* t1   = (float*)(ws + (((size_t)N * 4 + 255) & ~(size_t)255));
    float* t2   = t1 + (size_t)N * D_HID;

    const int TB = 256;
    const int nBlkN   = (N + TB - 1) / TB;
    const int nBlkE   = (E + TB - 1) / TB;
    const int nBlkEW  = (E * 32 + TB - 1) / TB;              // wave per edge
    const int nBlkND4 = (N * (D_HID / 4) + TB - 1) / TB;
    const int nBlkND  = (N * D_HID + TB - 1) / TB;
    const int nBlkGemm = ((N / 16) + 7) / 8;                 // 3125 tiles / 8 waves

    // 1) symmetric normalization: deg over dst (+self), dinv = rsqrt(deg)
    deg_init_kernel<<<nBlkN, TB, 0, stream>>>(dinv, N);
    deg_edge_kernel<<<nBlkE, TB, 0, stream>>>(dst, dinv, E);
    dinv_kernel<<<nBlkN, TB, 0, stream>>>(dinv, N);

    // 2) layer 1: t1 = x @ W1 ; t2 = norm-aggregate(t1) ; relu(t2 + b1)
    gcn_gemm_kernel<256, 8, false><<<nBlkGemm, TB, 0, stream>>>(x, W1, nullptr, t1, N);
    agg_init_kernel<<<nBlkND4, TB, 0, stream>>>(t1, dinv, t2, N, D_HID);
    agg_edge_kernel<<<nBlkEW, TB, 0, stream>>>(t1, src, dst, dinv, t2, E, D_HID);
    bias_relu_kernel<<<nBlkND, TB, 0, stream>>>(t2, b1, N * D_HID, D_HID);

    // 3) layer 2: t1 = t2 @ W2 ; t2 = norm-aggregate(t1) ; relu(t2 + b2)
    gcn_gemm_kernel<128, 8, false><<<nBlkGemm, TB, 0, stream>>>(t2, W2, nullptr, t1, N);
    agg_init_kernel<<<nBlkND4, TB, 0, stream>>>(t1, dinv, t2, N, D_HID);
    agg_edge_kernel<<<nBlkEW, TB, 0, stream>>>(t1, src, dst, dinv, t2, E, D_HID);
    bias_relu_kernel<<<nBlkND, TB, 0, stream>>>(t2, b2, N * D_HID, D_HID);

    // 4) head: out = t2 @ Wf + bf   [N, 64]
    gcn_gemm_kernel<128, 4, true><<<nBlkGemm, TB, 0, stream>>>(t2, Wf, bf, out, N);
}
